// PLinear_44719199486507
// MI455X (gfx1250) — matrix operand order repaired and checked by reading, not compiled, not run on hardware
//
#include <hip/hip_runtime.h>

typedef __attribute__((ext_vector_type(16))) __bf16 v16bf;
typedef __attribute__((ext_vector_type(8)))  __bf16 v8bf;
typedef __attribute__((ext_vector_type(4)))  __bf16 v4bf;
typedef __attribute__((ext_vector_type(8)))  float  v8f;
typedef __attribute__((ext_vector_type(4)))  float  f32x4;
typedef __attribute__((ext_vector_type(4)))  int    v4i;

// Address-space-qualified pointer types for the async-LDS builtin
typedef v4i __attribute__((address_space(1)))* gas_b128;   // global
typedef v4i __attribute__((address_space(3)))* las_b128;   // LDS

// Problem sizes (fixed by the reference)
constexpr int Mtot = 8 * 1024;   // B*S
constexpr int Ntot = 4096;       // D_OUT
constexpr int Ktot = 4096;       // D_IN

// Tiling: block = 128x256, 8 waves, each wave computes 64x64 (16 WMMA tiles)
constexpr int BM = 128, BN = 256, BK = 32;
constexpr int LDT = BK + 8;      // padded LDS row stride: 80B, keeps 16B alignment

#if __has_builtin(__builtin_amdgcn_global_load_async_to_lds_b128)
#define HAVE_ASYNC_LDS 1
#else
#define HAVE_ASYNC_LDS 0
#endif

__device__ __forceinline__ void copy16_g2l(const __bf16* g, __bf16* l) {
#if HAVE_ASYNC_LDS
    // gfx1250 async copy: global -> LDS, no VGPR round-trip, tracked by ASYNCcnt
    __builtin_amdgcn_global_load_async_to_lds_b128(
        (gas_b128)(g), (las_b128)(l), 0, 0);
#else
    *(f32x4*)l = *(const f32x4*)g;
#endif
}

__device__ __forceinline__ void wait_async_copies() {
#if HAVE_ASYNC_LDS
#if __has_builtin(__builtin_amdgcn_s_wait_asynccnt)
    __builtin_amdgcn_s_wait_asynccnt(0);
#else
    asm volatile("s_wait_asynccnt 0" ::: "memory");
#endif
#endif
}

__device__ __forceinline__ __bf16 f32_to_bf16(float f) {
    unsigned u = __builtin_bit_cast(unsigned, f);
    unsigned r = (u + 0x7FFFu + ((u >> 16) & 1u)) >> 16;   // round-to-nearest-even
    return __builtin_bit_cast(__bf16, (unsigned short)r);
}

// --- Kernel 1: x (f32) -> bf16, vectorized x4 ------------------------------
__global__ __launch_bounds__(256) void cvt_x_bf16(const float* __restrict__ x,
                                                  __bf16* __restrict__ xb) {
    size_t i = (size_t)blockIdx.x * blockDim.x + threadIdx.x;
    f32x4 v = __builtin_nontemporal_load((const f32x4*)x + i);  // one-shot stream
    v4bf o;
    o.x = f32_to_bf16(v.x); o.y = f32_to_bf16(v.y);
    o.z = f32_to_bf16(v.z); o.w = f32_to_bf16(v.w);
    ((v4bf*)xb)[i] = o;                       // keep resident: GEMM re-reads it
}

// --- Kernel 2: fold STE-quantized weights into one ternary bf16 matrix -----
// W[n,k] = (w_pos>0 ? 1 : 0) + (w_neg<=0 ? -1 : 0)  in {-1,0,1} (exact in bf16)
__global__ __launch_bounds__(256) void quant_w_ternary(const float* __restrict__ wp,
                                                       const float* __restrict__ wn,
                                                       __bf16* __restrict__ wb) {
    size_t i = (size_t)blockIdx.x * blockDim.x + threadIdx.x;
    f32x4 p = __builtin_nontemporal_load((const f32x4*)wp + i);
    f32x4 n = __builtin_nontemporal_load((const f32x4*)wn + i);
    v4bf o;
    o.x = f32_to_bf16((p.x > 0.f ? 1.f : 0.f) - (n.x <= 0.f ? 1.f : 0.f));
    o.y = f32_to_bf16((p.y > 0.f ? 1.f : 0.f) - (n.y <= 0.f ? 1.f : 0.f));
    o.z = f32_to_bf16((p.z > 0.f ? 1.f : 0.f) - (n.z <= 0.f ? 1.f : 0.f));
    o.w = f32_to_bf16((p.w > 0.f ? 1.f : 0.f) - (n.w <= 0.f ? 1.f : 0.f));
    ((v4bf*)wb)[i] = o;
}

// --- Kernel 3: bf16 WMMA GEMM  Y[m,n] = sum_k X[m,k] * W[n,k] --------------
__global__ __launch_bounds__(256) void ternary_gemm_wmma(
        const __bf16* __restrict__ X,   // [Mtot, Ktot] row-major bf16
        const __bf16* __restrict__ W,   // [Ntot, Ktot] row-major bf16
        float* __restrict__ Y) {        // [Mtot, Ntot] row-major f32
    __shared__ __bf16 As[2][BM * LDT];  // 2 x 10 KB
    __shared__ __bf16 Bs[2][BN * LDT];  // 2 x 20 KB

    const int tid  = threadIdx.x;
    const int lane = tid & 31;
    const int wid  = tid >> 5;      // 8 waves
    const int wM   = wid >> 2;      // 0..1 -> 64-row band
    const int wN   = wid & 3;       // 0..3 -> 64-col band

    const int mBase = blockIdx.y * BM;
    const int nBase = blockIdx.x * BN;

    // Cooperative staging assignment (per thread, per K-step):
    //   A tile 128x32: 2 x 16B   B tile 256x32: 4 x 16B
    const int arow  = tid >> 1;           // 0..127
    const int ahalf = (tid & 1) * 16;     // 0 or 16 elements
    const __bf16* gA = X + (size_t)(mBase + arow) * Ktot + ahalf;
    const int sA = arow * LDT + ahalf;

    const int brow = tid;                 // 0..255
    const __bf16* gB = W + (size_t)(nBase + brow) * Ktot;
    const int sB = brow * LDT;

    v8f acc[4][4] = {};

    // ---- prologue: stage K-tile 0 into buffer 0
    copy16_g2l(gA,      &As[0][sA]);
    copy16_g2l(gA + 8,  &As[0][sA + 8]);
    copy16_g2l(gB,      &Bs[0][sB]);
    copy16_g2l(gB + 8,  &Bs[0][sB + 8]);
    copy16_g2l(gB + 16, &Bs[0][sB + 16]);
    copy16_g2l(gB + 24, &Bs[0][sB + 24]);
    wait_async_copies();
    __syncthreads();

    const int nT = Ktot / BK;             // 128 K-steps
    const int rowInHalf = lane & 15;
    const int kbA = (lane & 16) ? 8  : 0; // A frag: lanes 16-31 hold K 8-15 / 24-31
    const int kbB = (lane & 16) ? 8  : 0; // B frag: lanes 16-31 hold K 8-15 / 24-31

    for (int t = 0; t < nT; ++t) {
        const int cur = t & 1, nxt = cur ^ 1;
        const bool pre = (t + 1 < nT);

        // issue next tile's async copies early (hide under WMMA);
        // LDS[nxt] was last read in iteration t-1, before its trailing barrier
        if (pre) {
            const __bf16* pa = gA + (size_t)(t + 1) * BK;
            const __bf16* pb = gB + (size_t)(t + 1) * BK;
            copy16_g2l(pa,      &As[nxt][sA]);
            copy16_g2l(pa + 8,  &As[nxt][sA + 8]);
            copy16_g2l(pb,      &Bs[nxt][sB]);
            copy16_g2l(pb + 8,  &Bs[nxt][sB + 8]);
            copy16_g2l(pb + 16, &Bs[nxt][sB + 16]);
            copy16_g2l(pb + 24, &Bs[nxt][sB + 24]);
        }

        // A fragments: 4 M-tiles per wave
        v16bf afrag[4];
#pragma unroll
        for (int i = 0; i < 4; ++i) {
            const __bf16* p = &As[cur][(wM * 64 + i * 16 + rowInHalf) * LDT];
            v8bf lo = *(const v8bf*)(p + kbA);
            v8bf hi = *(const v8bf*)(p + kbA + 16);
            afrag[i] = __builtin_shufflevector(lo, hi,
                0,1,2,3,4,5,6,7,8,9,10,11,12,13,14,15);
        }
        // B fragments: 4 N-tiles per wave
        v16bf bfrag[4];
#pragma unroll
        for (int j = 0; j < 4; ++j) {
            const __bf16* p = &Bs[cur][(wN * 64 + j * 16 + rowInHalf) * LDT + kbB];
            v8bf lo = *(const v8bf*)(p);
            v8bf hi = *(const v8bf*)(p + 16);
            bfrag[j] = __builtin_shufflevector(lo, hi,
                0,1,2,3,4,5,6,7,8,9,10,11,12,13,14,15);
        }

        // 16x v_wmma_f32_16x16x32_bf16 per wave per K-step
#pragma unroll
        for (int i = 0; i < 4; ++i)
#pragma unroll
            for (int j = 0; j < 4; ++j)
                acc[i][j] = __builtin_amdgcn_wmma_f32_16x16x32_bf16(
                    false, afrag[i], false, bfrag[j],
                    (short)0, acc[i][j], false, false);

        if (pre) wait_async_copies();   // own copies landed; barrier makes all visible
        __syncthreads();
    }

    // ---- epilogue: C layout VGPR r -> M = r (lanes 0-15) / r+8 (lanes 16-31)
    const int mHalf = (lane >> 4) << 3;
    const int col0  = lane & 15;
#pragma unroll
    for (int i = 0; i < 4; ++i)
#pragma unroll
        for (int j = 0; j < 4; ++j)
#pragma unroll
            for (int r = 0; r < 8; ++r) {
                int row = mBase + wM * 64 + i * 16 + mHalf + r;
                int col = nBase + wN * 64 + j * 16 + col0;
                // streaming output: non-temporal, keep X/W tiles in L2
                __builtin_nontemporal_store(acc[i][j][r],
                                            &Y[(size_t)row * Ntot + col]);
            }
}

extern "C" void kernel_launch(void* const* d_in, const int* in_sizes, int n_in,
                              void* d_out, int out_size, void* d_ws, size_t ws_size,
                              hipStream_t stream) {
    (void)in_sizes; (void)n_in; (void)out_size; (void)ws_size;
    const float* x  = (const float*)d_in[0];
    const float* wp = (const float*)d_in[1];
    const float* wn = (const float*)d_in[2];
    float* y = (float*)d_out;

    // workspace: x_bf16 (64 MB) then W_bf16 (32 MB)
    __bf16* xb = (__bf16*)d_ws;
    __bf16* wb = (__bf16*)((char*)d_ws + (size_t)Mtot * Ktot * sizeof(__bf16));

    {   // x: 33.5M elems, 4 per thread
        int blocks = (Mtot * Ktot / 4) / 256;
        cvt_x_bf16<<<blocks, 256, 0, stream>>>(x, xb);
    }
    {   // W: 16.8M elems, 4 per thread
        int blocks = (Ntot * Ktot / 4) / 256;
        quant_w_ternary<<<blocks, 256, 0, stream>>>(wp, wn, wb);
    }
    {
        dim3 grid(Ntot / BN, Mtot / BM);   // 16 x 64 blocks
        ternary_gemm_wmma<<<grid, 256, 0, stream>>>(xb, wb, y);
    }
}
